// Half_Graph_5420248727997
// MI455X (gfx1250) — matrix operand order repaired and checked by reading, not compiled, not run on hardware
//
#include <hip/hip_runtime.h>
#include <math.h>

// ---------------------------------------------------------------------------
// Half_Graph fused kernel for MI455X (gfx1250, wave32, WMMA bf16)
//
// One workgroup (256 threads = 8 wave32) per 16x16 spatial tile per batch
// image (2048 blocks). Memory-bound problem (~250 MB @ 23.3 TB/s ~= 11 us);
// the 8 fused 3x3 convs (20->10 ch) run on v_wmma_f32_16x16x32_bf16 with
// layout-native LDS fragments:
//   A frag = 2x ds_load_b128 from pixel-major sPix[pix][10 blk][16 ch bf16]
//   B frag = 2x ds_load_b128 from column-major sW[set][n][288]
// K per WMMA = one 3x3 tap: [blkA 10ch + 6 pad | blkB 10ch + 6 pad].
// ---------------------------------------------------------------------------

#define HID    10
#define BATCH  8
#define IMG    256
#define HWIMG  (IMG*IMG)
#define TILE   16
#define HALO   18
#define HPIX   (HALO*HALO)      // 324
#define PITCHE 168              // bf16 elems per staged pixel (336 B, 16B mult)
#define KW     288              // 9 taps * 32 padded K per weight column
#define NBLK   10               // staged channel blocks

// flat float offsets of the 5 concatenated outputs
#define O_XU  0
#define O_XL  5242880           // 8*10*65536
#define O_DM  10485760
#define O_CMU 12058624
#define O_CML 12582912

typedef __bf16 bf16_t;
typedef __bf16 v16bf __attribute__((ext_vector_type(16)));
typedef __bf16 v8bf  __attribute__((ext_vector_type(8)));
typedef float  v8f   __attribute__((ext_vector_type(8)));

union BF16x16 { v16bf v; v8bf h[2]; };

struct HGParams {
  const float *f, *h, *p;
  const float *da_w, *da_b;
  const float *dr_w, *dr_g, *dr_b, *dr_m, *dr_v;
  const float *cau_w, *cau_b, *cal_w, *cal_b;
  const float *cr_w, *cr_g, *cr_b, *cr_m, *cr_v;
  const float *gu_gw, *gu_gb, *gu_cw, *gu_cb;
  const float *gl_gw, *gl_gb, *gl_cw, *gl_cb;
  float *out;
};

__device__ __forceinline__ void gru_cell(const float* gw, const float* gb,
                                         const float* cw, const float* cb,
                                         const float* m_, const float* h_,
                                         float* outp /* stride HWIMG per ch */) {
  float r_[HID], u_[HID];
#pragma unroll
  for (int j = 0; j < 2*HID; ++j) {
    float a = gb[j];
#pragma unroll
    for (int k = 0; k < HID; ++k) a += gw[j*2*HID + k]       * m_[k];
#pragma unroll
    for (int k = 0; k < HID; ++k) a += gw[j*2*HID + HID + k] * h_[k];
    float s = 1.f / (1.f + __expf(-a));
    if (j < HID) r_[j] = s; else u_[j-HID] = s;
  }
#pragma unroll
  for (int j = 0; j < HID; ++j) {
    float a = cb[j];
#pragma unroll
    for (int k = 0; k < HID; ++k) a += cw[j*2*HID + k]       * m_[k];
#pragma unroll
    for (int k = 0; k < HID; ++k) a += cw[j*2*HID + HID + k] * (r_[k]*h_[k]);
    float cd = tanhf(a);
    outp[(size_t)j*HWIMG] = (1.f - u_[j]) * h_[j] + u_[j] * cd;
  }
}

__launch_bounds__(256)
__global__ void half_graph_fused(HGParams P) {
  // ---- LDS (147.8 KB total) ----
  __shared__ __align__(16) bf16_t sPix[HPIX * PITCHE];   // 108,864 B
  __shared__ __align__(16) bf16_t sW[2 * 16 * KW];       //  18,432 B
  __shared__ float sMsgU[HID * 256];                     //  10,240 B
  __shared__ float sMsgL[HID * 256];                     //  10,240 B

  const int tid  = threadIdx.x;
  const int blk  = blockIdx.x;
  const int b    = blk >> 8;              // 256 tiles per image
  const int trem = blk & 255;
  const int ty   = trem >> 4;
  const int tx   = trem & 15;
  const int y0   = ty*TILE - 1;           // halo origin
  const int x0   = tx*TILE - 1;

  const float* fB  = P.f + (size_t)b*HID*HWIMG;
  const float* h0B = P.h + (size_t)b*HID*HWIMG;
  const float* h1B = P.h + (size_t)(BATCH + b)*HID*HWIMG;

  // ---- stage weight panels, column-major: sW[set][n][k], k = tap*32 + kloc,
  //      kloc<16 -> concat-ch kloc (blockA), kloc>=16 -> concat-ch 10+kloc-16 ----
  for (int idx = tid; idx < 2*16*KW; idx += 256) {
    const int st = idx / (16*KW);
    const int r  = idx - st*(16*KW);
    const int n  = r / KW;
    const int k  = r - n*KW;
    const int q  = k >> 5;                // tap 0..8 = ky*3+kx
    const int kl = k & 31;
    float v = 0.f;
    const bool okA = (kl < 16) && (kl < HID);
    const bool okB = (kl >= 16) && (kl - 16 < HID);
    if (n < HID && (okA || okB)) {
      const int c  = okA ? kl : (10 + kl - 16);
      const float* w = st ? P.cr_w : P.dr_w;       // [O=10][I=20][3][3]
      v = w[((n*20 + c)*3 + (q/3))*3 + (q % 3)];
    }
    sW[idx] = (bf16_t)v;
  }

  // ---- stage halo pixels (pointwise attention in f32, bf16 products) ----
  for (int p = tid; p < HPIX; p += 256) {
    const int hy = p / HALO, hx = p - hy*HALO;
    const int gy = y0 + hy,  gx = x0 + hx;
    bf16_t* pixp = &sPix[p * PITCHE];
    if (!(gy >= 0 && gy < IMG && gx >= 0 && gx < IMG)) {
      const v8bf z = {};
#pragma unroll
      for (int i = 0; i < PITCHE/8; ++i) *(v8bf*)(pixp + i*8) = z;
      continue;
    }
    const int o = gy*IMG + gx;
    float fv[HID], h0v[HID], h1v[HID], pv[6][HID];
#pragma unroll
    for (int c = 0; c < HID; ++c) {
      fv[c]  = fB [c*HWIMG + o];
      h0v[c] = h0B[c*HWIMG + o];
      h1v[c] = h1B[c*HWIMG + o];
    }
    for (int i = 0; i < 6; ++i)
#pragma unroll
      for (int c = 0; c < HID; ++c)
        pv[i][c] = P.p[((size_t)i*BATCH + b)*HID*HWIMG + c*HWIMG + o];

    // decomp att: 1x1 conv (10->3) + softmax (channels 1,2 used)
    float dm[3];
#pragma unroll
    for (int oc = 0; oc < 3; ++oc) {
      float a = P.da_b[oc];
#pragma unroll
      for (int c = 0; c < HID; ++c) a += P.da_w[oc*HID + c]*fv[c];
      dm[oc] = a;
    }
    const float mx = fmaxf(dm[0], fmaxf(dm[1], dm[2]));
    const float e0 = __expf(dm[0]-mx), e1 = __expf(dm[1]-mx), e2 = __expf(dm[2]-mx);
    const float inv = 1.f/(e0+e1+e2);
    const float a1 = e1*inv, a2 = e2*inv;

    // comp att maps
    float su_ = P.cau_b[0], sl_ = P.cal_b[0];
    for (int i = 0; i < 4; ++i)
#pragma unroll
      for (int c = 0; c < HID; ++c) su_ += P.cau_w[i*HID + c]*pv[i][c];
    for (int i = 0; i < 2; ++i)
#pragma unroll
      for (int c = 0; c < HID; ++c) sl_ += P.cal_w[i*HID + c]*pv[4+i][c];
    const float cmu = 1.f/(1.f + __expf(-su_));
    const float cml = 1.f/(1.f + __expf(-sl_));

    // blocks: 0:f*a1 1:f*a2 2:h0 3:h1 4..7:p_i*cmu 8..9:p_i*cml
    float blkv[NBLK][HID];
#pragma unroll
    for (int c = 0; c < HID; ++c) {
      blkv[0][c] = fv[c]*a1;    blkv[1][c] = fv[c]*a2;
      blkv[2][c] = h0v[c];      blkv[3][c] = h1v[c];
      blkv[4][c] = pv[0][c]*cmu; blkv[5][c] = pv[1][c]*cmu;
      blkv[6][c] = pv[2][c]*cmu; blkv[7][c] = pv[3][c]*cmu;
      blkv[8][c] = pv[4][c]*cml; blkv[9][c] = pv[5][c]*cml;
    }
#pragma unroll
    for (int bk = 0; bk < NBLK; ++bk) {
      v8bf lo, hi;
#pragma unroll
      for (int c = 0; c < 8; ++c) lo[c] = (bf16_t)blkv[bk][c];
      hi[0] = (bf16_t)blkv[bk][8]; hi[1] = (bf16_t)blkv[bk][9];
#pragma unroll
      for (int c = 2; c < 8; ++c) hi[c] = (bf16_t)0.0f;
      *(v8bf*)(pixp + bk*16)     = lo;
      *(v8bf*)(pixp + bk*16 + 8) = hi;
    }

    // interior pixel: emit exact pointwise outputs
    if (hy >= 1 && hy <= TILE && hx >= 1 && hx <= TILE) {
#pragma unroll
      for (int oc = 0; oc < 3; ++oc)
        P.out[O_DM + ((size_t)b*3 + oc)*HWIMG + o] = dm[oc];
      P.out[O_CMU + (size_t)b*HWIMG + o] = cmu;
      P.out[O_CML + (size_t)b*HWIMG + o] = cml;
    }
  }
  __syncthreads();

  // ---- conv phase: 8 waves x 2 rows x 8 instances, 9 WMMAs each ----
  const int lane = tid & 31;
  const int wid  = tid >> 5;
  const int mN   = lane & 15;             // pixel-x (A rows) / out-ch (B cols)
  const int sel  = lane >> 4;
  const int nn   = (mN < HID) ? mN : 0;   // clamped BN-param index

  // hoisted BN affine params for both sets (single v_rsq each)
  const float scD = P.dr_g[nn] * rsqrtf(P.dr_v[nn] + 1e-5f);
  const float mnD = P.dr_m[nn], btD = P.dr_b[nn];
  const float scC = P.cr_g[nn] * rsqrtf(P.cr_v[nn] + 1e-5f);
  const float mnC = P.cr_m[nn], btC = P.cr_b[nn];

#pragma unroll 1
  for (int gi = 0; gi < 2; ++gi) {
    const int g = wid*2 + gi;             // tile row 0..15
    // A-frag row base (bf16 elems): lanes 0-15 take ch 0..7, 16-31 ch 8..15
    const int rowE = (g*HALO + mN)*PITCHE + sel*8;
    float su[8] = {0,0,0,0,0,0,0,0};
    float sl[8] = {0,0,0,0,0,0,0,0};

#pragma unroll 1
    for (int inst = 0; inst < 8; ++inst) {
      // concat blocks: inst0 [f*a1,h0] inst1 [f*a2,h1] inst2..5 [h0,Pu_i]
      //                inst6..7 [h1,Pl_i]
      const int blkA = (inst == 0) ? 0 : (inst == 1) ? 1 : (inst < 6) ? 2 : 3;
      const int blkB = inst + 2;
      const int st   = (inst < 2) ? 0 : 1;        // dr vs cr set
      const bf16_t* aA = &sPix[rowE + blkA*16];
      const bf16_t* aB = &sPix[rowE + blkB*16];
      const bf16_t* bb = &sW[(st*16 + mN)*KW + sel*8];

      v8f acc = {0.f,0.f,0.f,0.f,0.f,0.f,0.f,0.f};
#pragma unroll
      for (int q = 0; q < 9; ++q) {               // taps: q = ky*3+kx
        const int aoff = ((q/3)*HALO + (q%3)) * PITCHE;   // constant per q
        BF16x16 A, Bv;
        A.h[0]  = *(const v8bf*)(aA + aoff);      // VGPR0-3: blkA ch 8*sel..
        A.h[1]  = *(const v8bf*)(aB + aoff);      // VGPR4-7: blkB ch 8*sel..
        Bv.h[0] = *(const v8bf*)(bb + q*32);      // K = 8*sel + 0..7
        Bv.h[1] = *(const v8bf*)(bb + q*32 + 16); // K = 16 + 8*sel + 0..7
        acc = __builtin_amdgcn_wmma_f32_16x16x32_bf16(
                  false, A.v, false, Bv.v, (short)0, acc, false, false);
      }

      // BN + ReLU (params hoisted), accumulate into message sums
      const float scale = st ? scC : scD;
      const float mean  = st ? mnC : mnD;
      const float beta  = st ? btC : btD;
      const bool upper  = (inst == 0) || (inst >= 2 && inst <= 5);
#pragma unroll
      for (int j = 0; j < 8; ++j) {
        float r = (acc[j] - mean)*scale + beta;
        r = fmaxf(r, 0.f);
        if (upper) su[j] += r; else sl[j] += r;
      }
    }

    if (mN < HID) {
#pragma unroll
      for (int j = 0; j < 8; ++j) {
        const int px = g*16 + j + 8*sel;          // C-frag: VGPR j -> M=j+8*sel
        sMsgU[mN*256 + px] = su[j];
        sMsgL[mN*256 + px] = sl[j];
      }
    }
  }
  __syncthreads();

  // ---- GRU phase: one pixel per thread, f32 (h reread from global) ----
  {
    const int px = tid;
    const int gy = ty*TILE + (px >> 4);
    const int gx = tx*TILE + (px & 15);
    const int o  = gy*IMG + gx;
    float m_[HID], h_[HID];
#pragma unroll
    for (int c = 0; c < HID; ++c) { m_[c] = sMsgU[c*256+px]; h_[c] = h0B[c*HWIMG + o]; }
    gru_cell(P.gu_gw, P.gu_gb, P.gu_cw, P.gu_cb, m_, h_,
             P.out + O_XU + (size_t)b*HID*HWIMG + o);
#pragma unroll
    for (int c = 0; c < HID; ++c) { m_[c] = sMsgL[c*256+px]; h_[c] = h1B[c*HWIMG + o]; }
    gru_cell(P.gl_gw, P.gl_gb, P.gl_cw, P.gl_cb, m_, h_,
             P.out + O_XL + (size_t)b*HID*HWIMG + o);
  }
}

extern "C" void kernel_launch(void* const* d_in, const int* in_sizes, int n_in,
                              void* d_out, int out_size, void* d_ws, size_t ws_size,
                              hipStream_t stream) {
  (void)in_sizes; (void)n_in; (void)d_ws; (void)ws_size; (void)out_size;
  HGParams P;
  // setup_inputs() dict order; params in make_params() insertion order.
  P.f     = (const float*)d_in[0];   // f_node  [8,10,256,256]
  P.h     = (const float*)d_in[1];   // h_nodes [2,8,10,256,256]
  P.p     = (const float*)d_in[2];   // p_nodes [6,8,10,256,256]
  /* d_in[3] = xh (unused by reference) */
  P.da_w  = (const float*)d_in[4];
  P.da_b  = (const float*)d_in[5];
  P.dr_w  = (const float*)d_in[6];
  P.dr_g  = (const float*)d_in[7];
  P.dr_b  = (const float*)d_in[8];
  P.dr_m  = (const float*)d_in[9];
  P.dr_v  = (const float*)d_in[10];
  P.cau_w = (const float*)d_in[11];
  P.cau_b = (const float*)d_in[12];
  P.cal_w = (const float*)d_in[13];
  P.cal_b = (const float*)d_in[14];
  P.cr_w  = (const float*)d_in[15];
  P.cr_g  = (const float*)d_in[16];
  P.cr_b  = (const float*)d_in[17];
  P.cr_m  = (const float*)d_in[18];
  P.cr_v  = (const float*)d_in[19];
  P.gu_gw = (const float*)d_in[20];
  P.gu_gb = (const float*)d_in[21];
  P.gu_cw = (const float*)d_in[22];
  P.gu_cb = (const float*)d_in[23];
  P.gl_gw = (const float*)d_in[24];
  P.gl_gb = (const float*)d_in[25];
  P.gl_cw = (const float*)d_in[26];
  P.gl_cb = (const float*)d_in[27];
  P.out   = (float*)d_out;

  half_graph_fused<<<dim3(BATCH*256), dim3(256), 0, stream>>>(P);
}
